// GCNEdgeClassifier_45741401703144
// MI455X (gfx1250) — matrix operand order repaired and checked by reading, not compiled, not run on hardware
//
#include <hip/hip_runtime.h>
#include <hip/hip_bf16.h>

// ---------------------------------------------------------------------------
// GCN edge classifier forward for MI455X (gfx1250, wave32, WMMA).
// ---------------------------------------------------------------------------

typedef __attribute__((ext_vector_type(16))) _Float16 v16h;
typedef __attribute__((ext_vector_type(8)))  _Float16 v8h;
typedef __attribute__((ext_vector_type(8)))  float    v8f;

#define N_NODES 50000
#define N_EDGES 800000
#define IN_FEAT 4
#define HID     128

// ------------------------------- prep: weight convert + transpose ----------
// wg16t[l][n][k] = (f16) W_g_l[k*H + n]   (3 layers, 128x128)
// wm1t[n][k]     = (f16) W_m1[k*H + n]    (k in [0,256))
__global__ void prep_weights_kernel(const float* __restrict__ Wg1,
                                    const float* __restrict__ Wg2,
                                    const float* __restrict__ Wg3,
                                    const float* __restrict__ Wm1,
                                    _Float16* __restrict__ wg16t,
                                    _Float16* __restrict__ wm1t) {
    int idx = blockIdx.x * blockDim.x + threadIdx.x;
    const int HH = HID * HID;
    if (idx < 3 * HH) {
        int l = idx / HH;
        int rem = idx % HH;
        int n = rem / HID;
        int k = rem % HID;
        const float* W = (l == 0) ? Wg1 : (l == 1) ? Wg2 : Wg3;
        wg16t[idx] = (_Float16)W[k * HID + n];
    } else {
        int j = idx - 3 * HH;
        if (j < HID * 2 * HID) {
            int n = j / (2 * HID);
            int k = j % (2 * HID);
            wm1t[j] = (_Float16)Wm1[k * HID + n];
        }
    }
}

// ------------------------------- encoder: h = relu(x @ W_enc + b) ----------
__global__ void encoder_kernel(const float* __restrict__ x,
                               const float* __restrict__ Wenc,
                               const float* __restrict__ benc,
                               float* __restrict__ h32,
                               _Float16* __restrict__ h16) {
    int idx = blockIdx.x * blockDim.x + threadIdx.x;
    if (idx >= N_NODES * HID) return;
    int n = idx / HID;
    int f = idx % HID;
    float acc = benc[f];
    #pragma unroll
    for (int i = 0; i < IN_FEAT; ++i)
        acc += x[n * IN_FEAT + i] * Wenc[i * HID + f];
    float r = acc > 0.0f ? acc : 0.0f;
    h32[idx] = r;
    h16[idx] = (_Float16)r;
}

// ------------------------------- degree / norm -----------------------------
__global__ void deg_init_kernel(float* __restrict__ dinv) {
    int i = blockIdx.x * blockDim.x + threadIdx.x;
    if (i < N_NODES) dinv[i] = 1.0f;   // self-loop contribution
}
__global__ void deg_accum_kernel(const int* __restrict__ dst, float* __restrict__ dinv) {
    int e = blockIdx.x * blockDim.x + threadIdx.x;
    if (e < N_EDGES) atomicAdd(&dinv[dst[e]], 1.0f);
}
__global__ void deg_rsqrt_kernel(float* __restrict__ dinv) {
    int i = blockIdx.x * blockDim.x + threadIdx.x;
    if (i < N_NODES) dinv[i] = rsqrtf(dinv[i]);   // deg >= 1 always
}

// ------------------------------- WMMA helpers ------------------------------
__device__ inline v16h make_frag(v8h lo, v8h hi) {
    return __builtin_shufflevector(lo, hi, 0, 1, 2, 3, 4, 5, 6, 7,
                                           8, 9, 10, 11, 12, 13, 14, 15);
}
// A fragment (16x32 f16): lane holds row L%16; elems 0-7 = K (L/16)*8+{0..7},
// elems 8-15 = K 16+(L/16)*8+{0..7}  -> two contiguous 16B loads.
__device__ inline v16h load_a_frag(const _Float16* rowptr, int kb, int half) {
    v8h lo = *(const v8h*)(rowptr + kb + half * 8);
    v8h hi = *(const v8h*)(rowptr + kb + 16 + half * 8);
    return make_frag(lo, hi);
}
// B fragment (32x16 f16) from transposed weights Wt[n][k]: lane holds column
// L%16, elems 0-7 = K (L/16)*8+{0..7}, elems 8-15 = K 16+(L/16)*8+{0..7}.
__device__ inline v16h load_b_frag(const _Float16* nrowptr, int kb, int half) {
    v8h lo = *(const v8h*)(nrowptr + kb + half * 8);
    v8h hi = *(const v8h*)(nrowptr + kb + 16 + half * 8);
    return make_frag(lo, hi);
}

// ------------------------------- node GEMM: t = h16 @ W (K=H=N=128) --------
// One wave per PAIR of 16-row M tiles (32 rows): every B fragment is used by
// two wmmas, halving loads and load-wait stalls per matrix op. 64 wmma/wave.
__global__ __launch_bounds__(128) void node_gemm_kernel(
        const _Float16* __restrict__ A,    // [N_NODES][128] f16
        const _Float16* __restrict__ Bt,   // [128][128] f16 transposed [n][k]
        float* __restrict__ C) {           // [N_NODES][128] f32
    int wave = threadIdx.x >> 5;
    int lane = threadIdx.x & 31;
    int pair = blockIdx.x * 4 + wave;
    int t0   = pair * 2;
    if (t0 * 16 >= N_NODES) return;            // wave-uniform
    bool has2 = ((t0 + 1) * 16) < N_NODES;     // second tile valid?
    int m0   = t0 * 16;
    int m1   = m0 + 16;
    int nlo  = lane & 15;
    int half = lane >> 4;
    const _Float16* arow0 = A + (size_t)(m0 + nlo) * HID;
    const _Float16* arow1 = has2 ? A + (size_t)(m1 + nlo) * HID : arow0;

    v8f acc0[8] = {};
    v8f acc1[8] = {};
    #pragma unroll
    for (int kt = 0; kt < 4; ++kt) {
        int kb = kt * 32;
        v16h af0 = load_a_frag(arow0, kb, half);
        v16h af1 = load_a_frag(arow1, kb, half);
        #pragma unroll
        for (int g = 0; g < 2; ++g) {
            v16h bf[4];
            #pragma unroll
            for (int j = 0; j < 4; ++j)
                bf[j] = load_b_frag(Bt + (size_t)((g * 4 + j) * 16 + nlo) * HID, kb, half);
            #pragma unroll
            for (int j = 0; j < 4; ++j) {
                int nt = g * 4 + j;
                acc0[nt] = __builtin_amdgcn_wmma_f32_16x16x32_f16(
                    false, af0, false, bf[j], (short)0, acc0[nt], false, false);
                acc1[nt] = __builtin_amdgcn_wmma_f32_16x16x32_f16(
                    false, af1, false, bf[j], (short)0, acc1[nt], false, false);
            }
        }
    }
    #pragma unroll
    for (int nt = 0; nt < 8; ++nt) {
        int n = nt * 16 + nlo;
        #pragma unroll
        for (int r = 0; r < 8; ++r)
            C[(size_t)(m0 + r + half * 8) * HID + n] = acc0[nt][r];
    }
    if (has2) {
        #pragma unroll
        for (int nt = 0; nt < 8; ++nt) {
            int n = nt * 16 + nlo;
            #pragma unroll
            for (int r = 0; r < 8; ++r)
                C[(size_t)(m1 + r + half * 8) * HID + n] = acc1[nt][r];
        }
    }
}

// ------------------------------- init accumulator with bias ----------------
__global__ void init_acc_kernel(const float* __restrict__ b, float* __restrict__ hacc) {
    int idx = blockIdx.x * blockDim.x + threadIdx.x;
    if (idx < N_NODES * HID) hacc[idx] = b[idx % HID];
}

// ------------------------------- scatter messages --------------------------
// One wave per edge (incl. self-loops), 4 feats/lane, f32 atomics into hacc.
__global__ void scatter_kernel(const float* __restrict__ t,
                               const int* __restrict__ src,
                               const int* __restrict__ dst,
                               const float* __restrict__ dinv,
                               float* __restrict__ hacc) {
    int e = blockIdx.x * 8 + (threadIdx.x >> 5);
    if (e >= N_EDGES + N_NODES) return;
    int lane = threadIdx.x & 31;
    int s, d;
    if (e < N_EDGES) { s = src[e]; d = dst[e]; }
    else             { s = d = e - N_EDGES; }
    float w = dinv[s] * dinv[d];
    const float4 v = *(const float4*)(t + (size_t)s * HID + lane * 4);
    float* o = hacc + (size_t)d * HID + lane * 4;
    atomicAdd(o + 0, v.x * w);
    atomicAdd(o + 1, v.y * w);
    atomicAdd(o + 2, v.z * w);
    atomicAdd(o + 3, v.w * w);
}

// ------------------------------- relu + f16 refresh ------------------------
__global__ void relu_update_kernel(const float* __restrict__ hacc,
                                   float* __restrict__ h32,
                                   _Float16* __restrict__ h16) {
    int idx = blockIdx.x * blockDim.x + threadIdx.x;
    if (idx >= N_NODES * HID) return;
    float v = hacc[idx];
    float r = v > 0.0f ? v : 0.0f;
    h32[idx] = r;
    h16[idx] = (_Float16)r;
}

// ------------------------------- fused edge MLP ----------------------------
// Per wave: 32 edges (two 16-row M tiles sharing all B fragments).
// Gather [h[src]|h[dst]] (32x256 f16) into LDS, z = relu(ef @ W_m1 + b1)
// via 128 wmma (K=256, N=128), z tile (32x128 f32) in LDS,
// out = z @ W_m2 + b2 (128->3), one edge per lane.
__global__ __launch_bounds__(64) void edge_mlp_kernel(
        const _Float16* __restrict__ h16,
        const int* __restrict__ src,
        const int* __restrict__ dst,
        const _Float16* __restrict__ Wm1t,  // [128][256] f16 transposed [n][k]
        const float* __restrict__ b1,
        const float* __restrict__ Wm2,      // [128][3] f32
        const float* __restrict__ b2,
        float* __restrict__ out) {          // [N_EDGES][3] f32
    __shared__ __align__(16) _Float16 Atile[2][32][2 * HID]; // 32 KB
    __shared__ __align__(16) float    Ztile[2][32][HID];     // 32 KB
    int wave = threadIdx.x >> 5;
    int lane = threadIdx.x & 31;
    int tile = blockIdx.x * 2 + wave;       // 25000 tiles, grid=12500 exact
    int e0   = tile * 32;

    // Stage A: each lane gathers one edge: src row -> [0,128), dst -> [128,256)
    {
        int e  = e0 + lane;
        int ns = src[e];
        int nd = dst[e];
        const v8h* sp = (const v8h*)(h16 + (size_t)ns * HID);
        const v8h* dq = (const v8h*)(h16 + (size_t)nd * HID);
        v8h* q = (v8h*)(&Atile[wave][lane][0]);
        #pragma unroll
        for (int t = 0; t < 16; ++t) q[t] = sp[t];
        #pragma unroll
        for (int t = 0; t < 16; ++t) q[16 + t] = dq[t];
    }
    __syncthreads();

    int nlo  = lane & 15;
    int half = lane >> 4;
    const _Float16* arow0 = &Atile[wave][nlo][0];
    const _Float16* arow1 = &Atile[wave][16 + nlo][0];

    v8f acc0[8] = {};
    v8f acc1[8] = {};
    #pragma unroll
    for (int kt = 0; kt < 8; ++kt) {
        int kb = kt * 32;
        v16h af0 = load_a_frag(arow0, kb, half);
        v16h af1 = load_a_frag(arow1, kb, half);
        #pragma unroll
        for (int g = 0; g < 2; ++g) {
            v16h bf[4];
            #pragma unroll
            for (int j = 0; j < 4; ++j)
                bf[j] = load_b_frag(Wm1t + (size_t)((g * 4 + j) * 16 + nlo) * (2 * HID), kb, half);
            #pragma unroll
            for (int j = 0; j < 4; ++j) {
                int nt = g * 4 + j;
                acc0[nt] = __builtin_amdgcn_wmma_f32_16x16x32_f16(
                    false, af0, false, bf[j], (short)0, acc0[nt], false, false);
                acc1[nt] = __builtin_amdgcn_wmma_f32_16x16x32_f16(
                    false, af1, false, bf[j], (short)0, acc1[nt], false, false);
            }
        }
    }

    // bias + relu -> z tiles in LDS
    #pragma unroll
    for (int nt = 0; nt < 8; ++nt) {
        int n = nt * 16 + nlo;
        float bn = b1[n];
        #pragma unroll
        for (int r = 0; r < 8; ++r) {
            float z0 = acc0[nt][r] + bn;
            Ztile[wave][r + half * 8][n] = z0 > 0.0f ? z0 : 0.0f;
            float z1 = acc1[nt][r] + bn;
            Ztile[wave][16 + r + half * 8][n] = z1 > 0.0f ? z1 : 0.0f;
        }
    }
    __syncthreads();

    // head: 128 -> 3, one edge per lane
    {
        float o0 = b2[0], o1 = b2[1], o2 = b2[2];
        const float* z = &Ztile[wave][lane][0];
        #pragma unroll 4
        for (int f = 0; f < HID; ++f) {
            float zv = z[f];
            o0 += zv * Wm2[f * 3 + 0];
            o1 += zv * Wm2[f * 3 + 1];
            o2 += zv * Wm2[f * 3 + 2];
        }
        size_t e = (size_t)(e0 + lane);
        out[e * 3 + 0] = o0;
        out[e * 3 + 1] = o1;
        out[e * 3 + 2] = o2;
    }
}

// ---------------------------------------------------------------------------
extern "C" void kernel_launch(void* const* d_in, const int* in_sizes, int n_in,
                              void* d_out, int out_size, void* d_ws, size_t ws_size,
                              hipStream_t stream) {
    // inputs (setup_inputs order)
    const float* x     = (const float*)d_in[0];
    const int*   ei    = (const int*)d_in[1];      // [2][E]
    const float* Wenc  = (const float*)d_in[2];
    const float* benc  = (const float*)d_in[3];
    const float* Wg1   = (const float*)d_in[4];
    const float* bg1   = (const float*)d_in[5];
    const float* Wg2   = (const float*)d_in[6];
    const float* bg2   = (const float*)d_in[7];
    const float* Wg3   = (const float*)d_in[8];
    const float* bg3   = (const float*)d_in[9];
    const float* Wm1   = (const float*)d_in[10];
    const float* bm1   = (const float*)d_in[11];
    const float* Wm2   = (const float*)d_in[12];
    const float* bm2   = (const float*)d_in[13];
    const int* src = ei;
    const int* dst = ei + N_EDGES;
    float* out = (float*)d_out;

    // workspace carve (all 16B aligned by construction)
    const size_t NH = (size_t)N_NODES * HID;
    float*    h32   = (float*)d_ws;
    float*    t32   = h32 + NH;
    float*    hacc  = t32 + NH;
    float*    dinv  = hacc + NH;
    _Float16* h16   = (_Float16*)(dinv + N_NODES);
    _Float16* wg16t = h16 + NH;                    // 3*128*128
    _Float16* wm1t  = wg16t + 3 * HID * HID;       // 128*256

    const int TB = 256;
    // weight convert/transpose
    {
        int total = 3 * HID * HID + HID * 2 * HID;
        prep_weights_kernel<<<(total + TB - 1) / TB, TB, 0, stream>>>(
            Wg1, Wg2, Wg3, Wm1, wg16t, wm1t);
    }
    // encoder
    encoder_kernel<<<(int)((NH + TB - 1) / TB), TB, 0, stream>>>(x, Wenc, benc, h32, h16);
    // degree / normalization
    deg_init_kernel<<<(N_NODES + TB - 1) / TB, TB, 0, stream>>>(dinv);
    deg_accum_kernel<<<(N_EDGES + TB - 1) / TB, TB, 0, stream>>>(dst, dinv);
    deg_rsqrt_kernel<<<(N_NODES + TB - 1) / TB, TB, 0, stream>>>(dinv);

    // 3 GCN layers
    const float* biases[3] = {bg1, bg2, bg3};
    const int nPairs = (N_NODES / 16 + 1) / 2;     // 1563 tile-pairs
    const int gemmBlocks = (nPairs + 3) / 4;       // 391
    const int scatBlocks = (N_EDGES + N_NODES + 7) / 8;
    for (int l = 0; l < 3; ++l) {
        node_gemm_kernel<<<gemmBlocks, 128, 0, stream>>>(
            h16, wg16t + (size_t)l * HID * HID, t32);
        init_acc_kernel<<<(int)((NH + TB - 1) / TB), TB, 0, stream>>>(biases[l], hacc);
        scatter_kernel<<<scatBlocks, 256, 0, stream>>>(t32, src, dst, dinv, hacc);
        relu_update_kernel<<<(int)((NH + TB - 1) / TB), TB, 0, stream>>>(hacc, h32, h16);
    }

    // fused edge MLP: 25000 32-edge tiles, 2 waves/block -> 12500 blocks exact
    edge_mlp_kernel<<<(N_EDGES / 32) / 2, 64, 0, stream>>>(
        h16, src, dst, wm1t, bm1, Wm2, bm2, out);
}
